// PytorchAttention_38182259262143
// MI455X (gfx1250) — matrix operand order repaired and checked by reading, not compiled, not run on hardware
//
#include <hip/hip_runtime.h>

// ---------------------------------------------------------------------------
// CDNA5 / gfx1250 flash-attention + LM-head, bf16 WMMA (16x16x32, f32 acc)
// Round 2: 64-key tiles (half the softmax/barrier overhead per key),
//          global_prefetch of next K/V tile, NT stores for the 655MB output.
// ---------------------------------------------------------------------------

typedef __attribute__((ext_vector_type(16))) __bf16        v16bf;
typedef __attribute__((ext_vector_type(8)))  float         v8f;
typedef __attribute__((ext_vector_type(4)))  unsigned int  v4u;
typedef __attribute__((ext_vector_type(4)))  float         v4f;
typedef __attribute__((ext_vector_type(2)))  float         v2f;

#define D_MODEL 128
#define SEQ     4096
#define BATCH   4
#define VOCAB   10000

__device__ __forceinline__ unsigned short f2bf(float x) {
    unsigned int u = __float_as_uint(x);
    unsigned int r = u + 0x7FFFu + ((u >> 16) & 1u);   // round-to-nearest-even
    return (unsigned short)(r >> 16);
}
__device__ __forceinline__ unsigned int pack2bf(float lo, float hi) {
    return (unsigned int)f2bf(lo) | ((unsigned int)f2bf(hi) << 16);
}

union FragBF {                 // one 16x32 bf16 WMMA operand (8 VGPRs/lane)
    v16bf        bf;
    v4u          u4[2];
    unsigned int u[8];
};

__device__ __forceinline__ float rmax16(float x) {
    #pragma unroll
    for (int m = 1; m < 16; m <<= 1) x = fmaxf(x, __shfl_xor(x, m));
    return x;                  // masks < 16 never cross the 16-lane half
}
__device__ __forceinline__ float rsum16(float x) {
    #pragma unroll
    for (int m = 1; m < 16; m <<= 1) x += __shfl_xor(x, m);
    return x;
}

// ---------------------------------------------------------------------------
// Kernel 0: W_lm fp32 -> bf16 (stays L2-resident for the GEMM)
// ---------------------------------------------------------------------------
__global__ void cvt_w_kernel(const float* __restrict__ W,
                             unsigned short* __restrict__ Wb, int npairs) {
    int i = blockIdx.x * blockDim.x + threadIdx.x;
    if (i < npairs) {
        v2f w = *(const v2f*)(W + 2 * (size_t)i);
        ((unsigned int*)Wb)[i] = pack2bf(w.x, w.y);
    }
}

// ---------------------------------------------------------------------------
// Kernel 1: causal flash attention. Block = 128 thr = 4 waves; each wave owns
// a 16-query tile; WG owns 64 queries, iterating over 64-key tiles.
// K tile [64][128] bf16 row-major; V kept as two transposed [128][32] halves
// so every WMMA B-fragment is a pair of contiguous ds_load_b128.
// ---------------------------------------------------------------------------
__global__ __launch_bounds__(128)
void attn_kernel(const float* __restrict__ Q, const float* __restrict__ K,
                 const float* __restrict__ V, unsigned short* __restrict__ attn_bf)
{
    __shared__ __align__(16) unsigned short Kt[64 * 128];       // 16 KB
    __shared__ __align__(16) unsigned short VT[2][128 * 32];    // 16 KB
    __shared__ __align__(16) unsigned short Pt[4][16 * 64];     //  8 KB

    const int lane  = threadIdx.x & 31;
    const int wave  = threadIdx.x >> 5;
    const int half  = lane >> 4;       // which 16-lane group
    const int l16   = lane & 15;
    const int b     = blockIdx.y;
    const int qb_wg = blockIdx.x * 64;
    const int qb    = qb_wg + wave * 16;

    const size_t bbase = (size_t)b * SEQ * D_MODEL;

    // ---- load Q as four A-fragments (16x32 each), convert fp32->bf16 -------
    FragBF qfrag[4];
    {
        const float* qp = Q + bbase + (size_t)(qb + l16) * D_MODEL;
        #pragma unroll
        for (int s = 0; s < 4; ++s) {
            const int dbase = s * 32 + half * 8;         // lanes>=16 take K+8
            v4f a0 = *(const v4f*)(qp + dbase);
            v4f a1 = *(const v4f*)(qp + dbase + 4);
            v4f b0 = *(const v4f*)(qp + dbase + 16);     // K+16 block
            v4f b1 = *(const v4f*)(qp + dbase + 20);
            qfrag[s].u[0] = pack2bf(a0.x, a0.y);
            qfrag[s].u[1] = pack2bf(a0.z, a0.w);
            qfrag[s].u[2] = pack2bf(a1.x, a1.y);
            qfrag[s].u[3] = pack2bf(a1.z, a1.w);
            qfrag[s].u[4] = pack2bf(b0.x, b0.y);
            qfrag[s].u[5] = pack2bf(b0.z, b0.w);
            qfrag[s].u[6] = pack2bf(b1.x, b1.y);
            qfrag[s].u[7] = pack2bf(b1.z, b1.w);
        }
    }

    v8f   O[8];                         // 16x128 output accumulator (8 N-chunks)
    float m_run[8], l_run[8];
    #pragma unroll
    for (int c = 0; c < 8; ++c) {
        #pragma unroll
        for (int r = 0; r < 8; ++r) O[c][r] = 0.0f;
    }
    #pragma unroll
    for (int r = 0; r < 8; ++r) { m_run[r] = -__builtin_inff(); l_run[r] = 0.0f; }

    const int ktiles = ((qb_wg + 63) >> 6) + 1;   // causal bound for this WG
    const float scale = 0.08838834764831845f;     // 1/sqrt(128)

    for (int kt = 0; kt < ktiles; ++kt) {
        const int kb = kt * 64;

        __syncthreads();   // previous-iteration readers done before refill
        {   // K tile: [key][d] bf16, packed-pair stores (b32)
            const float* kp = K + bbase + (size_t)kb * D_MODEL;
            #pragma unroll 4
            for (int i = threadIdx.x; i < 4096; i += 128) {
                const int key = i >> 6, d2 = (i & 63) * 2;
                v2f kk = *(const v2f*)(kp + (size_t)key * D_MODEL + d2);
                *(unsigned int*)&Kt[key * 128 + d2] = pack2bf(kk.x, kk.y);
            }
            // V tile transposed into two 32-key halves: VT[h][d][key]
            const float* vp = V + bbase + (size_t)kb * D_MODEL;
            #pragma unroll 4
            for (int i = threadIdx.x; i < 8192; i += 128) {
                const int key = i >> 7, d = i & 127;
                VT[key >> 5][d * 32 + (key & 31)] =
                    f2bf(vp[(size_t)key * D_MODEL + d]);
            }
        }
        __syncthreads();

        // prefetch next tile's K/V while this tile is being computed
        if (kt + 1 < ktiles) {
            const float* nk = K + bbase + (size_t)(kb + 64) * D_MODEL;
            const float* nv = V + bbase + (size_t)(kb + 64) * D_MODEL;
            __builtin_prefetch(nk + threadIdx.x * 64, 0, 3);   // 128*256B = 32KB
            __builtin_prefetch(nv + threadIdx.x * 64, 0, 3);
        }

        if (kb <= qb + 15) {            // this wave still has unmasked keys
            // ---- scores: four 16x16 tiles, 4 chained WMMAs each (K over d) -
            v8f sc[4];
            #pragma unroll
            for (int t = 0; t < 4; ++t) {
                #pragma unroll
                for (int r = 0; r < 8; ++r) sc[t][r] = 0.0f;
                #pragma unroll
                for (int s = 0; s < 4; ++s) {
                    FragBF bf;   // B frag: lane = key column, K = d in table order
                    const unsigned short* kr =
                        &Kt[(t * 16 + l16) * 128 + s * 32 + half * 8];
                    bf.u4[0] = *(const v4u*)(kr);
                    bf.u4[1] = *(const v4u*)(kr + 16);
                    sc[t] = __builtin_amdgcn_wmma_f32_16x16x32_bf16(
                        false, qfrag[s].bf, false, bf.bf, (short)0, sc[t],
                        false, false);
                }
            }
            // ---- scale + causal mask (D layout: lane<->key, reg<->q row) ---
            #pragma unroll
            for (int t = 0; t < 4; ++t) {
                const int key = kb + t * 16 + l16;
                #pragma unroll
                for (int r = 0; r < 8; ++r) {
                    const int q = qb + r + half * 8;
                    float sv = sc[t][r] * scale;
                    sc[t][r] = (key > q) ? -__builtin_inff() : sv;
                }
            }
            // ---- online softmax (row reductions over 16-lane halves) -------
            float corr[8];
            #pragma unroll
            for (int r = 0; r < 8; ++r) {
                float mx = fmaxf(fmaxf(sc[0][r], sc[1][r]),
                                 fmaxf(sc[2][r], sc[3][r]));
                mx         = rmax16(mx);
                float mnew = fmaxf(m_run[r], mx);
                float c    = __expf(m_run[r] - mnew);
                float ps   = 0.0f;
                #pragma unroll
                for (int t = 0; t < 4; ++t) {
                    float p = __expf(sc[t][r] - mnew);
                    sc[t][r] = p;
                    ps += p;
                }
                l_run[r] = l_run[r] * c + rsum16(ps);
                m_run[r] = mnew;
                corr[r]  = c;
            }
            #pragma unroll
            for (int c = 0; c < 8; ++c) {
                #pragma unroll
                for (int r = 0; r < 8; ++r) O[c][r] *= corr[r];
            }
            // ---- P (D-layout) -> LDS [q][key(64)] bf16, reload as A frags --
            unsigned short* pw = &Pt[wave][0];
            #pragma unroll
            for (int t = 0; t < 4; ++t) {
                #pragma unroll
                for (int r = 0; r < 8; ++r)
                    pw[(r + half * 8) * 64 + t * 16 + l16] = f2bf(sc[t][r]);
            }
            FragBF pa[2];
            #pragma unroll
            for (int h = 0; h < 2; ++h) {
                const unsigned short* pr = pw + l16 * 64 + h * 32 + half * 8;
                pa[h].u4[0] = *(const v4u*)(pr);        // keys {0..7}|{8..15}
                pa[h].u4[1] = *(const v4u*)(pr + 16);   // keys {16..23}|{24..31}
            }
            // ---- O += P x V  (8 chunks of 16 d-columns, 2 key-halves) ------
            #pragma unroll
            for (int c = 0; c < 8; ++c) {
                #pragma unroll
                for (int h = 0; h < 2; ++h) {
                    FragBF vb;   // B frag from transposed V: lane = d column
                    const unsigned short* vr =
                        &VT[h][(c * 16 + l16) * 32 + half * 8];
                    vb.u4[0] = *(const v4u*)(vr);
                    vb.u4[1] = *(const v4u*)(vr + 16);
                    O[c] = __builtin_amdgcn_wmma_f32_16x16x32_bf16(
                        false, pa[h].bf, false, vb.bf, (short)0, O[c],
                        false, false);
                }
            }
        }
    }

    // ---- normalize and emit attn_out as bf16 to workspace ------------------
    #pragma unroll
    for (int r = 0; r < 8; ++r) l_run[r] = 1.0f / l_run[r];
    unsigned short* op = attn_bf + ((size_t)b * SEQ + qb) * D_MODEL;
    #pragma unroll
    for (int c = 0; c < 8; ++c) {
        const int d = c * 16 + l16;
        #pragma unroll
        for (int r = 0; r < 8; ++r) {
            const int q = r + half * 8;
            op[(size_t)q * D_MODEL + d] = f2bf(O[c][r] * l_run[r]);
        }
    }
}

// ---------------------------------------------------------------------------
// Kernel 2: LM head GEMM: out[16384,10000] = attn_bf16[16384,128] * W_bf16^T.
// Block = 256 thr = 8 waves; wave w owns M-tile blockIdx.x*8+w and an 80-wide
// vocab strip (5 N-subtiles). Store-bound (655 MB fp32 out) -> NT stores keep
// W_bf16 / attn_bf16 resident in the 192MB L2.
// ---------------------------------------------------------------------------
__global__ __launch_bounds__(256)
void lmhead_kernel(const unsigned short* __restrict__ A,
                   const unsigned short* __restrict__ W,
                   float* __restrict__ out)
{
    const int lane = threadIdx.x & 31;
    const int wave = threadIdx.x >> 5;
    const int half = lane >> 4;
    const int l16  = lane & 15;
    const int mb   = (blockIdx.x * 8 + wave) * 16;
    const int vb   = blockIdx.y * 80;

    FragBF a[4];
    {
        const unsigned short* ap = A + (size_t)(mb + l16) * D_MODEL;
        #pragma unroll
        for (int s = 0; s < 4; ++s) {
            const int dbase = s * 32 + half * 8;
            a[s].u4[0] = *(const v4u*)(ap + dbase);
            a[s].u4[1] = *(const v4u*)(ap + dbase + 16);
        }
    }

    v8f acc[5];
    #pragma unroll
    for (int n = 0; n < 5; ++n) {
        #pragma unroll
        for (int r = 0; r < 8; ++r) acc[n][r] = 0.0f;
    }

    #pragma unroll
    for (int s = 0; s < 4; ++s) {
        #pragma unroll
        for (int n = 0; n < 5; ++n) {
            FragBF bf;    // B frag: lane = vocab column, K = d in table order
            const unsigned short* wp =
                W + (size_t)(vb + n * 16 + l16) * D_MODEL + s * 32 + half * 8;
            bf.u4[0] = *(const v4u*)(wp);
            bf.u4[1] = *(const v4u*)(wp + 16);
            acc[n] = __builtin_amdgcn_wmma_f32_16x16x32_bf16(
                false, a[s].bf, false, bf.bf, (short)0, acc[n], false, false);
        }
    }

    #pragma unroll
    for (int n = 0; n < 5; ++n) {
        const int v = vb + n * 16 + l16;
        #pragma unroll
        for (int r = 0; r < 8; ++r) {
            const int m = mb + r + half * 8;
            __builtin_nontemporal_store(acc[n][r], &out[(size_t)m * VOCAB + v]);
        }
    }
}

// ---------------------------------------------------------------------------
extern "C" void kernel_launch(void* const* d_in, const int* in_sizes, int n_in,
                              void* d_out, int out_size, void* d_ws, size_t ws_size,
                              hipStream_t stream) {
    const float* Q  = (const float*)d_in[0];
    const float* K  = (const float*)d_in[1];
    const float* V  = (const float*)d_in[2];
    const float* W  = (const float*)d_in[3];
    float*       out = (float*)d_out;

    // workspace: [0, 4 MB) attn_out bf16 ; [4 MB, 6.56 MB) W_lm bf16
    unsigned short* attn_bf = (unsigned short*)d_ws;
    unsigned short* W_bf    = attn_bf + (size_t)BATCH * SEQ * D_MODEL;

    const int wpairs = VOCAB * D_MODEL / 2;                     // 640000
    cvt_w_kernel<<<(wpairs + 255) / 256, 256, 0, stream>>>(W, W_bf, wpairs);

    attn_kernel<<<dim3(SEQ / 64, BATCH), 128, 0, stream>>>(Q, K, V, attn_bf);

    lmhead_kernel<<<dim3((BATCH * SEQ) / (16 * 8), VOCAB / 80), 256, 0, stream>>>(
        attn_bf, W_bf, out);
}
